// Model_24524263260167
// MI455X (gfx1250) — matrix operand order repaired and checked by reading, not compiled, not run on hardware
//
#include <hip/hip_runtime.h>
#include <hip/hip_bf16.h>
#include <stdint.h>

typedef unsigned short u16;
typedef __attribute__((ext_vector_type(16))) __bf16 v16bf;
typedef __attribute__((ext_vector_type(8)))  __bf16 v8bf;
typedef __attribute__((ext_vector_type(8)))  float  v8f;

#define DEVI __device__ __forceinline__

DEVI u16 f2bf(float f) {
  unsigned u = __builtin_bit_cast(unsigned, f);
  unsigned r = (u + 0x7FFFu + ((u >> 16) & 1u)) >> 16;   // round-to-nearest-even
  return (u16)r;
}

// ---------------------------------------------------------------------------
// Spherical-harmonic magnitude features (f32, channel-major [nm][c2][t][v]).
// feat[nm][sh*25+w][t][v] = K(l,|m|)*|P(l,|m|)(c,s)| of rel = pos[v]-pos[w].
// ---------------------------------------------------------------------------
__global__ __launch_bounds__(256) void k_sph(const float* __restrict__ x,
                                             float* __restrict__ feat)
{
  __shared__ float sx[25], sy[25];
  int bid = blockIdx.x;
  int nm = bid / 128;
  int t  = bid - nm * 128;
  int n = nm >> 1, m = nm & 1;
  if (threadIdx.x < 25) {
    int v = threadIdx.x;
    sx[v] = x[((((size_t)n*3 + 0)*128 + t)*25 + v)*2 + m];
    sy[v] = x[((((size_t)n*3 + 1)*128 + t)*25 + v)*2 + m];
  }
  __syncthreads();
  const float K10=0.48860251190292f,  K11=0.34549414947134f;
  const float K20=0.63078313050504f,  K21=0.25751613468213f, K22=0.12875806734106f;
  const float K30=0.74635266518023f,  K31=0.21545345607610f, K32=0.06813236509555f, K33=0.02781492157551f;
  for (int p = threadIdx.x; p < 625; p += blockDim.x) {
    int v = p % 25, w = p / 25;
    float xc = sx[v] - sx[w];
    float yc = sy[v] - sy[w];
    float r2 = xc*xc + yc*yc;
    float c, s;
    if (r2 > 0.0f) { float inv = rsqrtf(r2); c = xc*inv; s = fabsf(yc)*inv; }
    else           { c = 1.0f; s = 0.0f; }
    float ac = fabsf(c), s2 = s*s, c2 = c*c;
    float mg[15];
    mg[0]  = K11*s;                 mg[1]  = K10*ac;            mg[2]  = mg[0];
    mg[3]  = K22*3.0f*s2;           mg[4]  = K21*3.0f*ac*s;
    mg[5]  = K20*fabsf(0.5f*(3.0f*c2 - 1.0f));
    mg[6]  = mg[4];                 mg[7]  = mg[3];
    mg[8]  = K33*15.0f*s2*s;        mg[9]  = K32*15.0f*ac*s2;
    mg[10] = K31*1.5f*fabsf(5.0f*c2 - 1.0f)*s;
    mg[11] = K30*fabsf(0.5f*(5.0f*c2 - 3.0f)*c);
    mg[12] = mg[10];                mg[13] = mg[9];             mg[14] = mg[8];
    size_t base = ((size_t)nm*375)*3200 + (size_t)t*25 + v;
    #pragma unroll
    for (int sh = 0; sh < 15; ++sh)
      feat[base + (size_t)(sh*25 + w)*3200] = mg[sh];
  }
}

// ---------------------------------------------------------------------------
// data_bn statistics: channel ch = (m*25+v)*375 + c2, reduced over n(32)*t(128).
// ---------------------------------------------------------------------------
__global__ __launch_bounds__(256) void k_databn_stats(const float* __restrict__ feat,
                                                      float* __restrict__ mean,
                                                      float* __restrict__ rstd)
{
  __shared__ float ss[256], ss2[256];
  int ch = blockIdx.x;
  int c2 = ch % 375;
  int v  = (ch / 375) % 25;
  int m  = ch / (375*25);
  float s = 0.f, s2 = 0.f;
  for (int j = threadIdx.x; j < 32*128; j += 256) {
    int n = j >> 7, t = j & 127;
    float x = feat[(((size_t)(n*2 + m))*375 + c2)*3200 + t*25 + v];
    s += x; s2 += x*x;
  }
  ss[threadIdx.x] = s; ss2[threadIdx.x] = s2; __syncthreads();
  for (int st = 128; st > 0; st >>= 1) {
    if (threadIdx.x < st) { ss[threadIdx.x] += ss[threadIdx.x+st]; ss2[threadIdx.x] += ss2[threadIdx.x+st]; }
    __syncthreads();
  }
  if (threadIdx.x == 0) {
    float mu = ss[0] * (1.0f/4096.0f);
    float var = ss2[0] * (1.0f/4096.0f) - mu*mu;
    mean[ch] = mu; rstd[ch] = rsqrtf(fmaxf(var, 0.f) + 1e-5f);
  }
}

// Apply data-BN; bf16 output is POSITION-MAJOR: fb[(nm*3200+off)*384 + c2].
__global__ __launch_bounds__(256) void k_databn_apply(const float* __restrict__ feat,
    u16* __restrict__ fb, const float* __restrict__ mean, const float* __restrict__ rstd,
    const float* __restrict__ g, const float* __restrict__ b)
{
  long long idx = (long long)blockIdx.x*256 + threadIdx.x;
  if (idx >= 76800000LL) return;
  int off = (int)(idx % 3200);               // t*25 + v
  int c2  = (int)((idx / 3200) % 375);
  int nm  = (int)(idx / (3200LL*375));
  int v   = off % 25;
  int m   = nm & 1;
  int ch  = (m*25 + v)*375 + c2;
  float y = (feat[idx] - mean[ch]) * rstd[ch] * g[ch] + b[ch];
  fb[((size_t)nm*3200 + off)*384 + c2] = f2bf(y);
}

// Zero the 9 pad channels (375..383) of every position in featbf.
__global__ __launch_bounds__(256) void k_zero_pad(u16* __restrict__ fb)
{
  int idx = blockIdx.x*256 + threadIdx.x;
  if (idx >= 64*3200*9) return;
  int pos = idx / 9, c = 375 + idx % 9;
  fb[(size_t)pos*384 + c] = 0;
}

// ---------------------------------------------------------------------------
// Adjacency normalization: Anorm[k,v,w] = A[k,v,w] / (||A[k,:,w]||_2 + 1e-4)
// ---------------------------------------------------------------------------
__global__ void k_anorm(const float* __restrict__ PA, float* __restrict__ an)
{
  int tid = threadIdx.x;
  if (tid < 75) {
    int kk = tid / 25, w = tid % 25;
    float s = 0.f;
    for (int v = 0; v < 25; ++v) { float a = PA[kk*625 + v*25 + w]; s += a*a; }
    float inv = 1.0f / (sqrtf(s) + 1e-4f);
    for (int v = 0; v < 25; ++v) an[kk*625 + v*25 + w] = PA[kk*625 + v*25 + w] * inv;
  }
}

// ---------------------------------------------------------------------------
// Weight packing: f32 (O,C,taps) -> bf16 [O][taps*Cpad], k = dt*Cpad + c,
// zeros for c >= C (matches zero-padded activation channels).
// ---------------------------------------------------------------------------
__global__ __launch_bounds__(256) void k_pack_w(const float* __restrict__ W,
    u16* __restrict__ Wp, int O, int C, int taps, int Cpad)
{
  int Kpad = taps*Cpad;
  int idx = blockIdx.x*256 + threadIdx.x;
  if (idx >= O*Kpad) return;
  int o = idx / Kpad, k = idx - o*Kpad;
  int dt = k / Cpad, c = k - dt*Cpad;
  float v = (c < C) ? W[((size_t)o*C + c)*taps + dt] : 0.f;
  Wp[idx] = f2bf(v);
}

// ---------------------------------------------------------------------------
// WMMA bf16 implicit-GEMM over position-major activations.
//   Y[nm][o][to][v] = sum_{dt,c} Wp[o][dt*Cpad+c] * X[(nm,ti,v)][c]
//   X layout: [pos][Cpad], ti = to*STRIDE + dt - PAD.
// Per K-step: A = 2x b128, B = 2x b128 (two 16-byte runs, 32 bytes apart), 1 wmma.
// Pad-invalid taps: load clamped row, zero fragment by per-lane select.
// ---------------------------------------------------------------------------
template<int C, int TAPS, int STRIDE, int PAD, int TIN, int TOUT>
__global__ __launch_bounds__(128) void k_gemm_wmma(
    const u16* __restrict__ Wp, const u16* __restrict__ Xb,
    float* __restrict__ Y, int O, int Ntiles)
{
  constexpr int TVIN  = TIN * 25;
  constexpr int TVOUT = TOUT * 25;
  constexpr int CPAD  = (C + 31) & ~31;

  int wid  = blockIdx.x*4 + (threadIdx.x >> 5);
  int lane = threadIdx.x & 31;
  int Mtiles = O >> 4;
  if (wid >= Mtiles*Ntiles) return;       // tile counts are multiples of 4: never taken
  int mt = wid % Mtiles, nt = wid / Mtiles;
  int m0 = mt << 4, n0 = nt << 4;
  int nc = lane & 15, khalf = lane >> 4;
  int pos   = n0 + nc;
  int chunk = pos / TVOUT;
  int poff  = pos - chunk*TVOUT;
  int to    = poff / 25;
  int v     = poff - to*25;

  const u16* Arow = Wp + (size_t)(m0 + nc) * (TAPS*CPAD) + khalf*8;
  v8f acc = {0.f,0.f,0.f,0.f,0.f,0.f,0.f,0.f};

  #pragma unroll
  for (int dt = 0; dt < TAPS; ++dt) {
    int ti = to*STRIDE + dt - PAD;
    bool valid = true;
    if (PAD > 0) { valid = (unsigned)ti < (unsigned)TIN; if (!valid) ti = 0; }
    const u16* brow = Xb + ((size_t)chunk*TVIN + ti*25 + v) * CPAD + khalf*8;
    const u16* arow = Arow + dt*CPAD;
    for (int cs = 0; cs < CPAD/32; ++cs) {
      int cb = cs*32;
      v8bf alo = *(const v8bf*)(arow + cb);
      v8bf ahi = *(const v8bf*)(arow + cb + 16);
      v16bf a = __builtin_shufflevector(alo, ahi, 0,1,2,3,4,5,6,7,8,9,10,11,12,13,14,15);
      v8bf blo = *(const v8bf*)(brow + cb);
      v8bf bhi = *(const v8bf*)(brow + cb + 16);
      v16bf b = __builtin_shufflevector(blo, bhi, 0,1,2,3,4,5,6,7,8,9,10,11,12,13,14,15);
      if (PAD > 0 && !valid) { v16bf z = {}; b = z; }
      acc = __builtin_amdgcn_wmma_f32_16x16x32_bf16(false, a, false, b, (short)0, acc, false, false);
    }
  }
  size_t ybase = (size_t)chunk*O*TVOUT + poff;
  #pragma unroll
  for (int r = 0; r < 8; ++r)
    Y[ybase + (size_t)(m0 + khalf*8 + r)*TVOUT] = acc[r];
}

// ---------------------------------------------------------------------------
// Vertex mix: y[nm,o,t,w] (+)= sum_v z[nm,o,t,v] * Anorm[k][v][w]
// ---------------------------------------------------------------------------
__global__ __launch_bounds__(256) void k_vmix(const float* __restrict__ z,
    float* __restrict__ y, const float* __restrict__ an, int total, int accum)
{
  int idx = blockIdx.x*256 + threadIdx.x;
  if (idx >= total) return;
  int w = idx % 25;
  int row = idx / 25;
  const float* zr = z + (size_t)row*25;
  float s = 0.f;
  #pragma unroll
  for (int v = 0; v < 25; ++v) s += zr[v] * an[v*25 + w];
  if (accum) y[idx] += s; else y[idx] = s;
}

// ---------------------------------------------------------------------------
// Per-channel batch-norm statistics over (NM, T, V).
// ---------------------------------------------------------------------------
__global__ __launch_bounds__(256) void k_bn_stats(const float* __restrict__ x,
    float* __restrict__ mean, float* __restrict__ rstd, int C, int NMv, int TV)
{
  __shared__ float ss[256], ss2[256];
  int ch = blockIdx.x;
  int total = NMv*TV;
  float s = 0.f, s2 = 0.f;
  for (int i = threadIdx.x; i < total; i += 256) {
    int nm = i / TV, off = i - nm*TV;
    float v = x[((size_t)nm*C + ch)*TV + off];
    s += v; s2 += v*v;
  }
  ss[threadIdx.x] = s; ss2[threadIdx.x] = s2; __syncthreads();
  for (int st = 128; st > 0; st >>= 1) {
    if (threadIdx.x < st) { ss[threadIdx.x] += ss[threadIdx.x+st]; ss2[threadIdx.x] += ss2[threadIdx.x+st]; }
    __syncthreads();
  }
  if (threadIdx.x == 0) {
    float mu = ss[0] / total;
    float var = ss2[0] / total - mu*mu;
    mean[ch] = mu; rstd[ch] = rsqrtf(fmaxf(var, 0.f) + 1e-5f);
  }
}

// gcn epilogue: g = relu(bn(y) + (down? bn(resd) : x)); bf16 out position-major.
__global__ __launch_bounds__(256) void k_gcn_apply(
    const float* __restrict__ y, const float* __restrict__ mg, const float* __restrict__ rg,
    const float* __restrict__ gam, const float* __restrict__ bet,
    const float* __restrict__ resd, const float* __restrict__ md, const float* __restrict__ rd,
    const float* __restrict__ dgam, const float* __restrict__ dbet,
    const float* __restrict__ xf, int hasDown,
    u16* __restrict__ gout, int total, int TV, int C)
{
  int idx = blockIdx.x*256 + threadIdx.x;
  if (idx >= total) return;
  int nm   = idx / (C*TV);
  int rest = idx - nm*C*TV;
  int o    = rest / TV;
  int off  = rest - o*TV;
  float val = (y[idx] - mg[o]) * rg[o] * gam[o] + bet[o];
  float rv  = hasDown ? (resd[idx] - md[o]) * rd[o] * dgam[o] + dbet[o] : xf[idx];
  float out = fmaxf(val + rv, 0.0f);
  gout[((size_t)nm*TV + off)*C + o] = f2bf(out);
}

// tcn epilogue: out = relu(bn(yt) + r); f32 channel-major + bf16 position-major.
__global__ __launch_bounds__(256) void k_tcn_apply(
    const float* __restrict__ yt, const float* __restrict__ mt, const float* __restrict__ rt,
    const float* __restrict__ gam, const float* __restrict__ bet,
    const float* __restrict__ xf,
    const float* __restrict__ rres, const float* __restrict__ mr, const float* __restrict__ rrs,
    const float* __restrict__ rgam, const float* __restrict__ rbet,
    int mode, float* __restrict__ outf, u16* __restrict__ outbf, int total, int TV, int C)
{
  int idx = blockIdx.x*256 + threadIdx.x;
  if (idx >= total) return;
  int nm   = idx / (C*TV);
  int rest = idx - nm*C*TV;
  int o    = rest / TV;
  int off  = rest - o*TV;
  float val = (yt[idx] - mt[o]) * rt[o] * gam[o] + bet[o];
  float r = 0.0f;
  if (mode == 1)      r = xf[idx];
  else if (mode == 2) r = (rres[idx] - mr[o]) * rrs[o] * rgam[o] + rbet[o];
  float out = fmaxf(val + r, 0.0f);
  outf[idx] = out;
  outbf[((size_t)nm*TV + off)*C + o] = f2bf(out);
}

// ---------------------------------------------------------------------------
// Global average pool over (M, T*V) then FC 256 -> 60.
// ---------------------------------------------------------------------------
__global__ __launch_bounds__(256) void k_pool_fc(const float* __restrict__ act,
    const float* __restrict__ fcw, const float* __restrict__ fcb, float* __restrict__ out)
{
  __shared__ float pooled[256];
  int n = blockIdx.x;
  int cc = threadIdx.x;
  const int TV = 32*25;
  float s = 0.f;
  for (int m = 0; m < 2; ++m) {
    const float* base = act + (((size_t)(n*2 + m))*256 + cc)*TV;
    for (int j = 0; j < TV; ++j) s += base[j];
  }
  pooled[cc] = s * (1.0f/(2.0f*TV));
  __syncthreads();
  if (cc < 60) {
    float acc = fcb[cc];
    for (int j = 0; j < 256; ++j) acc += pooled[j] * fcw[cc*256 + j];
    out[n*60 + cc] = acc;
  }
}

// ---------------------------------------------------------------------------
// Host orchestration
// ---------------------------------------------------------------------------
extern "C" void kernel_launch(void* const* d_in, const int* in_sizes, int n_in,
                              void* d_out, int out_size, void* d_ws, size_t ws_size,
                              hipStream_t stream)
{
  (void)n_in; (void)out_size; (void)ws_size;
  static const int ci_[10] = {375,64,64,64,64,128,128,128,256,256};
  static const int co_[10] = {64,64,64,64,128,128,128,256,256,256};
  static const int st_[10] = {1,1,1,1,2,1,1,2,1,1};
  static const int rs_[10] = {0,1,1,1,1,1,1,1,1,1};

  struct LP { const float *PA,*gw,*gb,*gbg,*gbb,*tw,*tb,*tbg,*tbb,
                          *dw,*db,*dg,*dbt,*rw,*rb,*rbg,*rbb; };
  LP lp[10] = {};
  const float *X=nullptr, *dbg_=nullptr, *dbb_=nullptr, *fcw=nullptr, *fcb=nullptr;
  auto F = [&](int i){ return (const float*)d_in[i]; };

  // Insertion-order pytree (x first) vs key-sorted pytree (params leaves first).
  bool insertion = (in_sizes[0] == 32*3*128*25*2);
  int ix = 0;
  if (insertion) {
    X = F(ix++);
    for (int i = 0; i < 10; ++i) {
      bool hd = ci_[i] != co_[i];
      bool hr = rs_[i] && (ci_[i] != co_[i] || st_[i] != 1);
      lp[i].PA=F(ix++); lp[i].gw=F(ix++); lp[i].gb=F(ix++); lp[i].gbg=F(ix++); lp[i].gbb=F(ix++);
      lp[i].tw=F(ix++); lp[i].tb=F(ix++); lp[i].tbg=F(ix++); lp[i].tbb=F(ix++);
      if (hd) { lp[i].dw=F(ix++); lp[i].db=F(ix++); lp[i].dg=F(ix++); lp[i].dbt=F(ix++); }
      if (hr) { lp[i].rw=F(ix++); lp[i].rb=F(ix++); lp[i].rbg=F(ix++); lp[i].rbb=F(ix++); }
    }
    dbg_=F(ix++); dbb_=F(ix++); fcw=F(ix++); fcb=F(ix++);
  } else {
    dbb_=F(ix++); dbg_=F(ix++); fcb=F(ix++); fcw=F(ix++);
    for (int i = 0; i < 10; ++i) {
      bool hd = ci_[i] != co_[i];
      bool hr = rs_[i] && (ci_[i] != co_[i] || st_[i] != 1);
      lp[i].PA=F(ix++);
      if (hd) { lp[i].db=F(ix++); lp[i].dbt=F(ix++); lp[i].dg=F(ix++); lp[i].dw=F(ix++); }
      lp[i].gbb=F(ix++); lp[i].gbg=F(ix++); lp[i].gb=F(ix++); lp[i].gw=F(ix++);
      if (hr) { lp[i].rbb=F(ix++); lp[i].rbg=F(ix++); lp[i].rb=F(ix++); lp[i].rw=F(ix++); }
      lp[i].tbb=F(ix++); lp[i].tbg=F(ix++); lp[i].tb=F(ix++); lp[i].tw=F(ix++);
    }
    X = F(ix++);
  }

  // -------- workspace carve --------
  char* wp_ = (char*)d_ws;
  auto carve = [&](size_t bytes)->char* { char* r = wp_; wp_ += (bytes + 255) & ~(size_t)255; return r; };
  float* feat   = (float*)carve((size_t)76800000*4);   // f32 features; reused as z/y after data-BN
  u16*   featbf = (u16*)  carve((size_t)78643200*2);   // [pos][384] bf16
  float* zbuf   = feat;
  float* ybuf   = feat + 26214400;
  float* resb   = (float*)carve((size_t)26214400*4);
  float* ytb    = (float*)carve((size_t)13107200*4);
  u16*   gbf    = (u16*)  carve((size_t)26214400*2);   // [pos][co] bf16
  float* actAf  = (float*)carve((size_t)13107200*4);
  float* actBf  = (float*)carve((size_t)13107200*4);
  u16*   actAbf = (u16*)  carve((size_t)13107200*2);
  u16*   actBbf = (u16*)  carve((size_t)13107200*2);
  u16*   wpack  = (u16*)  carve((size_t)256*1280*2);
  float* anorm  = (float*)carve(1875*4);
  float* m_db   = (float*)carve(18750*4);
  float* r_db   = (float*)carve(18750*4);
  float* stats  = (float*)carve(8*256*4);
  float *mg=stats, *rg=stats+256, *md=stats+512, *rd=stats+768,
        *mt=stats+1024, *rt=stats+1280, *mr=stats+1536, *rr=stats+1792;

  // Specialized GEMM dispatch: 11 geometries cover the whole network.
  auto launch_gemm = [&](int C, int taps, int s, int Tin,
                         const u16* Wp, const u16* Xb, float* Y, int O) {
    int Tout = (s == 2) ? Tin/2 : Tin;
    int Nt = 64*Tout*25/16;
    dim3 g(((O/16)*Nt + 3)/4), b(128);
    #define GCASE(CC,TT,SS,TI) \
      if (C==CC && taps==TT && s==SS && Tin==TI) { \
        k_gemm_wmma<CC,TT,SS,((TT)==5?2:0),TI,((SS)==2?(TI)/2:(TI))> \
            <<<g, b, 0, stream>>>(Wp, Xb, Y, O, Nt); return; }
    GCASE(375,1,1,128) GCASE(64,1,1,128) GCASE(128,1,1,64) GCASE(256,1,1,32)
    GCASE(64,1,2,128)  GCASE(128,1,2,64)
    GCASE(64,5,1,128)  GCASE(128,5,2,128) GCASE(128,5,1,64)
    GCASE(256,5,2,64)  GCASE(256,5,1,32)
    #undef GCASE
  };

  // -------- features + data BN --------
  k_sph<<<dim3(64*128), dim3(256), 0, stream>>>(X, feat);
  k_databn_stats<<<dim3(18750), dim3(256), 0, stream>>>(feat, m_db, r_db);
  k_zero_pad<<<dim3((64*3200*9 + 255)/256), dim3(256), 0, stream>>>(featbf);
  k_databn_apply<<<dim3(300000), dim3(256), 0, stream>>>(feat, featbf, m_db, r_db, dbg_, dbb_);

  // -------- blocks --------
  int Tcur = 128;
  const u16* xbf = featbf;
  const float* xf = nullptr;
  for (int i = 0; i < 10; ++i) {
    int ci = ci_[i], co = co_[i], s = st_[i];
    int Tin = Tcur, Tout = (s == 2) ? Tcur/2 : Tcur;
    bool hd = ci != co;
    bool hr = rs_[i] && (ci != co || s != 1);
    int mode = !rs_[i] ? 0 : ((ci == co && s == 1) ? 1 : 2);
    int Cp1 = (ci + 31) & ~31;

    k_anorm<<<dim3(1), dim3(128), 0, stream>>>(lp[i].PA, anorm);
    for (int kk = 0; kk < 3; ++kk) {
      int nw = co*Cp1;
      k_pack_w<<<dim3((nw+255)/256), dim3(256), 0, stream>>>(lp[i].gw + (size_t)kk*co*ci, wpack, co, ci, 1, Cp1);
      launch_gemm(ci, 1, 1, Tin, wpack, xbf, zbuf, co);
      int tot = 64*co*Tin*25;
      k_vmix<<<dim3((tot+255)/256), dim3(256), 0, stream>>>(zbuf, ybuf, anorm + kk*625, tot, kk);
    }
    k_bn_stats<<<dim3(co), dim3(256), 0, stream>>>(ybuf, mg, rg, co, 64, Tin*25);
    if (hd) {
      int nw = co*Cp1;
      k_pack_w<<<dim3((nw+255)/256), dim3(256), 0, stream>>>(lp[i].dw, wpack, co, ci, 1, Cp1);
      launch_gemm(ci, 1, 1, Tin, wpack, xbf, resb, co);
      k_bn_stats<<<dim3(co), dim3(256), 0, stream>>>(resb, md, rd, co, 64, Tin*25);
    }
    {
      int tot = 64*co*Tin*25;
      k_gcn_apply<<<dim3((tot+255)/256), dim3(256), 0, stream>>>(ybuf, mg, rg, lp[i].gbg, lp[i].gbb,
          resb, md, rd, lp[i].dg, lp[i].dbt, xf, hd ? 1 : 0, gbf, tot, Tin*25, co);
    }
    {
      int nw = co*5*co;
      k_pack_w<<<dim3((nw+255)/256), dim3(256), 0, stream>>>(lp[i].tw, wpack, co, co, 5, co);
      launch_gemm(co, 5, s, Tin, wpack, gbf, ytb, co);
      k_bn_stats<<<dim3(co), dim3(256), 0, stream>>>(ytb, mt, rt, co, 64, Tout*25);
    }
    if (hr) {
      int nw = co*Cp1;
      k_pack_w<<<dim3((nw+255)/256), dim3(256), 0, stream>>>(lp[i].rw, wpack, co, ci, 1, Cp1);
      launch_gemm(ci, 1, s, Tin, wpack, xbf, resb, co);
      k_bn_stats<<<dim3(co), dim3(256), 0, stream>>>(resb, mr, rr, co, 64, Tout*25);
    }
    float* of = (i & 1) ? actBf : actAf;
    u16*  obf = (i & 1) ? actBbf : actAbf;
    {
      int tot = 64*co*Tout*25;
      k_tcn_apply<<<dim3((tot+255)/256), dim3(256), 0, stream>>>(ytb, mt, rt, lp[i].tbg, lp[i].tbb,
          xf, resb, mr, rr, lp[i].rbg, lp[i].rbb, mode, of, obf, tot, Tout*25, co);
    }
    xf = of; xbf = obf; Tcur = Tout;
  }

  k_pool_fc<<<dim3(32), dim3(256), 0, stream>>>(xf, fcw, fcb, (float*)d_out);
}
